// TrackerArray_84232898609678
// MI455X (gfx1250) — matrix operand order repaired and checked by reading, not compiled, not run on hardware
//
#include <hip/hip_runtime.h>
#include <hip/hip_bf16.h>

// ---------------------------------------------------------------------------
// NTM tracker recurrence for MI455X (gfx1250, wave32).
//
// grid = 4 blocks x 1024 threads (32 waves). Each block owns 16 batch
// elements; every weight matvec becomes an M=16 GEMM on
// v_wmma_f32_16x16x32_bf16 (one WMMA tile row). Whole T x O recurrence runs
// inside one kernel (batch partitions independent). C (256x128 per n) is
// streamed through L2 from d_ws scratch; activations live in LDS (~236 KB of
// the 320 KB WGP pool). A-operand activations are mirrored in LDS as bf16
// (converted once per phase) so WMMA fragment loads are raw ds_load_b128 with
// no per-tile conversion VALU. Weights pre-converted to bf16 in d_ws.
// Scratch: 1 MB (bf16 weights) + 8 MB (C) = 9 MB.
// ---------------------------------------------------------------------------

#define NB   64
#define TT   20
#define OO   4
#define HWm  256
#define FF   128
#define HH   256
#define GG   768   // 3*H
#define NPB  16    // batch elements per block
#define DIMY 136   // 1+3+4+64+64

// flat output segment offsets (fp32 elements)
#define OUT_HO 0ull
#define OUT_YE (OUT_HO + (size_t)NB*TT*OO*HH)
#define OUT_YL (OUT_YE + (size_t)NB*TT*OO*1)
#define OUT_YP (OUT_YL + (size_t)NB*TT*OO*3)
#define OUT_YS (OUT_YP + (size_t)NB*TT*OO*4)
#define OUT_YA (OUT_YS + (size_t)NB*TT*OO*64)

// bf16 weight pack offsets inside d_ws (ushort elements)
#define WOFF_WK  0
#define WOFF_WE  32768
#define WOFF_WV  65536
#define WOFF_WIH 98304
#define WOFF_WHH 196608
#define WOFF_WF  393216
#define WPACK_TOTAL 428032

typedef __attribute__((ext_vector_type(16))) __bf16 v16bf;
typedef __attribute__((ext_vector_type(8)))  float  v8f;

union BFrag { unsigned short us[16]; v16bf v; };

__device__ __forceinline__ unsigned short f2bf(float f) {
  unsigned int u = __float_as_uint(f);
  u += 0x7FFFu + ((u >> 16) & 1u);          // round-to-nearest-even
  return (unsigned short)(u >> 16);
}
__device__ __forceinline__ float sigmoidf_(float x) { return 1.f / (1.f + __expf(-x)); }
__device__ __forceinline__ float softplusf_(float x) { return (x > 20.f) ? x : log1pf(__expf(x)); }

// One 16x16 output tile of  Out(16 x Nout) = A(16 x KTOT) * W^T + bias.
// Abf: LDS bf16 (ushort) row-major, 16 x KTOT. W: global bf16 (NoutPhys x KTOT)
// row-major (so B = W^T). Executed by one full wave; EXEC all ones
// (wave-uniform call sites only). A fragment halves are contiguous 16B runs.
template <int KTOT>
__device__ __forceinline__ void gemm_tile16(const unsigned short* __restrict__ Abf,
                                            const unsigned short* __restrict__ W,
                                            const float* __restrict__ bias,
                                            float* __restrict__ Out, int ldOut,
                                            int tileN, int noutPhys) {
  const int lane  = threadIdx.x & 31;
  const int hi    = lane >> 4;            // 0: lanes 0-15, 1: lanes 16-31
  const int arow  = lane & 15;            // A row (M)
  const int koffA = hi << 3;              // A K offset per ISA layout
  const int koffB = hi << 4;              // unused; B loads use koffA (same k as A)
  const int col   = lane & 15;            // D column (N)
  const int gcol  = tileN * 16 + col;
  const bool colOK = gcol < noutPhys;

  float bval = (bias != nullptr && colOK) ? bias[gcol] : 0.f;
  v8f c;
#pragma unroll
  for (int i = 0; i < 8; ++i) c[i] = bval;

  const unsigned short* wrow = colOK ? (W + (size_t)gcol * KTOT) : W;  // OOB guard
  const unsigned short* arowp = Abf + arow * KTOT;
#pragma unroll
  for (int kb = 0; kb < KTOT; kb += 32) {
    BFrag a, b;
#pragma unroll
    for (int j = 0; j < 8; ++j) {
      a.us[j]     = arowp[kb + koffA + j];        // K = kb+koffA+0..7   (16B run)
      a.us[8 + j] = arowp[kb + koffA + 16 + j];   // K = kb+koffA+16..23 (16B run)
    }
#pragma unroll
    for (int j = 0; j < 8; ++j) {
      unsigned short wv0 = wrow[kb + koffA + j];
      unsigned short wv1 = wrow[kb + koffA + 16 + j];
      b.us[j]     = colOK ? wv0 : (unsigned short)0;
      b.us[8 + j] = colOK ? wv1 : (unsigned short)0;
    }
    c = __builtin_amdgcn_wmma_f32_16x16x32_bf16(false, a.v, false, b.v,
                                                (short)0, c, false, false);
  }
#pragma unroll
  for (int i = 0; i < 8; ++i) {
    int row = i + (hi << 3);
    if (colOK) Out[row * ldOut + gcol] = c[i];
  }
}

// ---------------------------------------------------------------------------
__global__ void prep_weights(const float* __restrict__ Wk, const float* __restrict__ We,
                             const float* __restrict__ Wv, const float* __restrict__ Wih,
                             const float* __restrict__ Whh, const float* __restrict__ Wf,
                             unsigned short* __restrict__ dst) {
  for (int i = blockIdx.x * blockDim.x + threadIdx.x; i < WPACK_TOTAL;
       i += gridDim.x * blockDim.x) {
    float v;
    if      (i < WOFF_WE)  v = Wk[i];
    else if (i < WOFF_WV)  v = We[i - WOFF_WE];
    else if (i < WOFF_WIH) v = Wv[i - WOFF_WV];
    else if (i < WOFF_WHH) v = Wih[i - WOFF_WIH];
    else if (i < WOFF_WF)  v = Whh[i - WOFF_WHH];
    else                   v = Wf[i - WOFF_WF];
    dst[i] = f2bf(v);
  }
}

// ---------------------------------------------------------------------------
__global__ __launch_bounds__(1024, 1)
void tracker_main(const float* __restrict__ h0,     // (N,O,H)
                  const float* __restrict__ feats,  // (N,T,HW,F)
                  const float* __restrict__ bk, const float* __restrict__ Wb,
                  const float* __restrict__ bb, const float* __restrict__ be,
                  const float* __restrict__ bv, const float* __restrict__ bih,
                  const float* __restrict__ bhh, const float* __restrict__ bfv,
                  const unsigned short* __restrict__ Wbf,
                  float* __restrict__ Cscr,         // (N,HW,F) fp32 scratch
                  float* __restrict__ out) {
  __shared__ alignas(16) float h_lds[OO][NPB][HH];           // 64 KB fp32 hidden
  __shared__ alignas(16) unsigned short h_bf[OO][NPB][HH];   // 32 KB bf16 mirror
  __shared__ alignas(16) unsigned short r_bf[NPB][FF];       // 4 KB bf16 read vec
  __shared__ alignas(16) float k_lds[NPB][FF];               // 8 KB
  __shared__ alignas(16) float s_lds[NPB][HWm];              // 16 KB (s, then w)
  __shared__ alignas(16) float gi_lds[NPB][GG];              // 48 KB (reused: a_lds 64x144)
  __shared__ alignas(16) float gh_lds[NPB][GG];              // 48 KB
  __shared__ alignas(16) float e_lds[NPB][FF];               // 8 KB
  __shared__ alignas(16) float v_lds[NPB][FF];               // 8 KB
  __shared__ float beta_lds[NPB];
  __shared__ float knorm_lds[NPB];

  const int tid   = threadIdx.x;
  const int wid   = tid >> 5;
  const int lane  = tid & 31;
  const int nbase = blockIdx.x * NPB;

  // load initial hidden state h_o_prev (fp32 + bf16 mirror)
  for (int idx = tid; idx < OO * NPB * HH; idx += 1024) {
    int o = idx / (NPB * HH);
    int i = (idx / HH) % NPB;
    int j = idx % HH;
    float hv = h0[((size_t)(nbase + i) * OO + o) * HH + j];
    h_lds[o][i][j] = hv;
    h_bf[o][i][j]  = f2bf(hv);
  }

  for (int t = 0; t < TT; ++t) {
    for (int o = 0; o < OO; ++o) {
      __syncthreads();
      // -- Phase A: k = h @ Wk^T + bk (WMMA, 8 tiles); beta (waves 16..31) --
      if (wid < 8) {
        gemm_tile16<HH>(&h_bf[o][0][0], Wbf + WOFF_WK, bk, &k_lds[0][0], FF, wid, FF);
      } else if (wid >= 16) {
        int n = wid - 16;
        float sum = 0.f;
        for (int j = lane; j < HH; j += 32) sum += h_lds[o][n][j] * Wb[j];
        for (int off = 16; off > 0; off >>= 1) sum += __shfl_xor(sum, off, 32);
        if (lane == 0) beta_lds[n] = softplusf_(sum + bb[0]) + 1.f;
      }
      __syncthreads();
      // -- Phase B: ||k|| per n --
      if (wid < 16) {
        int n = wid;
        float sum = 0.f;
        for (int j = lane; j < FF; j += 32) { float kv = k_lds[n][j]; sum += kv * kv; }
        for (int off = 16; off > 0; off >>= 1) sum += __shfl_xor(sum, off, 32);
        if (lane == 0) knorm_lds[n] = sqrtf(sum);
      }
      __syncthreads();
      // -- Phase C: cosine-sim scores s[n][m] (C streamed from feats / Cscr) --
      for (int idx = tid; idx < NPB * HWm; idx += 1024) {
        int n = idx >> 8;
        int m = idx & 255;
        const float* src = (o == 0)
            ? feats + ((size_t)((nbase + n) * TT + t) * HWm + m) * FF
            : Cscr + ((size_t)(nbase + n) * HWm + m) * FF;
        const float4* c4p = (const float4*)src;
        const float4* k4p = (const float4*)&k_lds[n][0];
        float dot = 0.f, ss = 0.f;
#pragma unroll 8
        for (int q = 0; q < FF / 4; ++q) {
          float4 c4 = c4p[q], k4 = k4p[q];
          dot += c4.x * k4.x + c4.y * k4.y + c4.z * k4.z + c4.w * k4.w;
          ss  += c4.x * c4.x + c4.y * c4.y + c4.z * c4.z + c4.w * c4.w;
        }
        float denom = fmaxf(sqrtf(ss) * knorm_lds[n], 1e-8f);
        s_lds[n][m] = (dot / denom) * beta_lds[n];
      }
      __syncthreads();
      // -- Phase D: softmax over m (one wave per n) --
      if (wid < 16) {
        int n = wid;
        float mx = -3.4e38f;
        for (int m = lane; m < HWm; m += 32) mx = fmaxf(mx, s_lds[n][m]);
        for (int off = 16; off > 0; off >>= 1) mx = fmaxf(mx, __shfl_xor(mx, off, 32));
        float sum = 0.f;
        for (int m = lane; m < HWm; m += 32) {
          float ev = __expf(s_lds[n][m] - mx);
          s_lds[n][m] = ev;
          sum += ev;
        }
        for (int off = 16; off > 0; off >>= 1) sum += __shfl_xor(sum, off, 32);
        float inv = 1.f / sum;
        for (int m = lane; m < HWm; m += 32) s_lds[n][m] *= inv;
      }
      __syncthreads();
      // -- Phase E: r[n][f] = sum_m w[n][m]*C[n][m][f], stored as bf16 --
      for (int idx = tid; idx < NPB * FF; idx += 1024) {
        int n = idx >> 7;
        int f = idx & 127;
        const float* src = (o == 0)
            ? feats + ((size_t)((nbase + n) * TT + t) * HWm) * FF + f
            : Cscr + ((size_t)(nbase + n) * HWm) * FF + f;
        float acc = 0.f;
        for (int m = 0; m < HWm; ++m) acc += s_lds[n][m] * src[(size_t)m * FF];
        r_bf[n][f] = f2bf(acc);
      }
      __syncthreads();
      // -- Phase F: GRU gate GEMMs (WMMA): gi = r@Wih^T+bih, gh = h@Whh^T+bhh --
      for (int vt = wid; vt < 96; vt += 32) {
        if (vt < 48)
          gemm_tile16<FF>(&r_bf[0][0], Wbf + WOFF_WIH, bih, &gi_lds[0][0], GG, vt, GG);
        else
          gemm_tile16<HH>(&h_bf[o][0][0], Wbf + WOFF_WHH, bhh, &gh_lds[0][0], GG, vt - 48, GG);
      }
      __syncthreads();
      // -- Phase G: GRU combine; write h_new (fp32 + bf16) and h_o_seq output --
      for (int idx = tid; idx < NPB * HH; idx += 1024) {
        int n = idx >> 8;
        int j = idx & 255;
        float rg = sigmoidf_(gi_lds[n][j] + gh_lds[n][j]);
        float zg = sigmoidf_(gi_lds[n][HH + j] + gh_lds[n][HH + j]);
        float ng = tanhf(gi_lds[n][2 * HH + j] + rg * gh_lds[n][2 * HH + j]);
        float hn = (1.f - zg) * ng + zg * h_lds[o][n][j];
        h_lds[o][n][j] = hn;
        h_bf[o][n][j]  = f2bf(hn);
        out[OUT_HO + (((size_t)(nbase + n) * TT + t) * OO + o) * HH + j] = hn;
      }
      if (o < OO - 1) {  // C after last object is discarded by the reference
        __syncthreads();
        // -- Phase H: e = sigmoid(h_new@We^T+be), v = h_new@Wv^T+bv (WMMA) --
        for (int vt = wid; vt < 16; vt += 32) {
          if (vt < 8)
            gemm_tile16<HH>(&h_bf[o][0][0], Wbf + WOFF_WE, be, &e_lds[0][0], FF, vt, FF);
          else
            gemm_tile16<HH>(&h_bf[o][0][0], Wbf + WOFF_WV, bv, &v_lds[0][0], FF, vt - 8, FF);
        }
        __syncthreads();
        // -- Phase I: sigmoid on e --
        for (int idx = tid; idx < NPB * FF; idx += 1024) {
          int n = idx >> 7, f = idx & 127;
          e_lds[n][f] = sigmoidf_(e_lds[n][f]);
        }
        __syncthreads();
        // -- Phase J: C update -> Cscr --
        for (int idx = tid; idx < NPB * HWm * (FF / 4); idx += 1024) {
          int n   = idx / (HWm * (FF / 4));
          int rem = idx % (HWm * (FF / 4));
          int m = rem / (FF / 4);
          int q = rem % (FF / 4);
          const float4* srcp = (o == 0)
              ? (const float4*)(feats + ((size_t)((nbase + n) * TT + t) * HWm + m) * FF) + q
              : (const float4*)(Cscr + ((size_t)(nbase + n) * HWm + m) * FF) + q;
          float4 c4 = *srcp;
          float w = s_lds[n][m];
          float4 e4 = *((const float4*)&e_lds[n][0] + q);
          float4 v4 = *((const float4*)&v_lds[n][0] + q);
          float4 r4;
          r4.x = c4.x * (1.f - w * e4.x) + w * v4.x;
          r4.y = c4.y * (1.f - w * e4.y) + w * v4.y;
          r4.z = c4.z * (1.f - w * e4.z) + w * v4.z;
          r4.w = c4.w * (1.f - w * e4.w) + w * v4.w;
          *((float4*)(Cscr + ((size_t)(nbase + n) * HWm + m) * FF) + q) = r4;
        }
      }
    }  // o

    __syncthreads();
    // -- Phase K: heads a = h_o @ Wf^T + bf -> a_lds (reuses gi_lds, 64x144) --
    float* a_lds = &gi_lds[0][0];
    for (int vt = wid; vt < 36; vt += 32) {
      int oo = vt / 9, tn = vt % 9;
      gemm_tile16<HH>(&h_bf[oo][0][0], Wbf + WOFF_WF, bfv,
                      a_lds + (size_t)oo * 16 * 144, 144, tn, DIMY);
    }
    __syncthreads();
    // -- Phase L: head outputs (16 threads per (n,o) row) --
    {
      int row = tid >> 4;       // 0..63 = oo*16 + i
      int l16 = tid & 15;
      int oo = row >> 4;
      int i  = row & 15;
      const float* ar = a_lds + (size_t)row * 144;
      size_t idx = ((size_t)(nbase + i) * TT + t) * OO + oo;
      if (l16 == 0) out[OUT_YE + idx] = ar[0];
      if (l16 == 1) {                       // hard one-hot argmax over 3 layers
        float m0 = ar[1]; int b = 0;
        if (ar[2] > m0) { m0 = ar[2]; b = 1; }
        if (ar[3] > m0) { m0 = ar[3]; b = 2; }
        out[OUT_YL + idx * 3 + 0] = (b == 0) ? 1.f : 0.f;
        out[OUT_YL + idx * 3 + 1] = (b == 1) ? 1.f : 0.f;
        out[OUT_YL + idx * 3 + 2] = (b == 2) ? 1.f : 0.f;
      }
      if (l16 == 2)
        for (int q = 0; q < 4; ++q) out[OUT_YP + idx * 4 + q] = ar[4 + q];
      for (int f = l16; f < 64; f += 16) {
        out[OUT_YS + idx * 64 + f] = (ar[8 + f] > 0.f) ? 1.f : 0.f;  // sigmoid>0.5
        out[OUT_YA + idx * 64 + f] = ar[72 + f];
      }
    }
  }  // t
}

// ---------------------------------------------------------------------------
extern "C" void kernel_launch(void* const* d_in, const int* in_sizes, int n_in,
                              void* d_out, int out_size, void* d_ws, size_t ws_size,
                              hipStream_t stream) {
  (void)in_sizes; (void)n_in; (void)out_size; (void)ws_size;
  const float* h0    = (const float*)d_in[0];
  // d_in[1] = y_e_prev (unused by reference)
  const float* feats = (const float*)d_in[2];
  const float* Wk    = (const float*)d_in[3];
  const float* bk    = (const float*)d_in[4];
  const float* Wb    = (const float*)d_in[5];
  const float* bb    = (const float*)d_in[6];
  const float* We    = (const float*)d_in[7];
  const float* be    = (const float*)d_in[8];
  const float* Wv    = (const float*)d_in[9];
  const float* bv    = (const float*)d_in[10];
  const float* Wih   = (const float*)d_in[11];
  const float* Whh   = (const float*)d_in[12];
  const float* bih   = (const float*)d_in[13];
  const float* bhh   = (const float*)d_in[14];
  const float* Wf    = (const float*)d_in[15];
  const float* bfv   = (const float*)d_in[16];

  unsigned short* wbf = (unsigned short*)d_ws;                  // 856,064 B
  float* Cscr = (float*)((char*)d_ws + (1u << 20));             // 8 MB at +1 MB

  prep_weights<<<418, 1024, 0, stream>>>(Wk, We, Wv, Wih, Whh, Wf, wbf);
  tracker_main<<<NB / NPB, 1024, 0, stream>>>(h0, feats, bk, Wb, bb, be, bv,
                                              bih, bhh, bfv, wbf, Cscr,
                                              (float*)d_out);
}
